// RNNDecoder_19447611916531
// MI455X (gfx1250) — matrix operand order repaired and checked by reading, not compiled, not run on hardware
//
#include <hip/hip_runtime.h>
#include <hip/hip_bf16.h>
#include <math.h>

#define B_  32
#define S_  128
#define H_  1024
#define V_  32000
#define BS_ (B_*S_)

typedef __attribute__((ext_vector_type(16))) __bf16 v16bf;
typedef __attribute__((ext_vector_type(8)))  float  v8f;

union Frag {
    v16bf v;
    uint4 q[2];
};

__device__ __forceinline__ unsigned short f2bf(float f) {
    unsigned int u = __float_as_uint(f);
    u += 0x7FFFu + ((u >> 16) & 1u);   // round-to-nearest-even
    return (unsigned short)(u >> 16);
}

__device__ __forceinline__ float sigmoidf_(float x) { return 1.0f / (1.0f + expf(-x)); }

// Async copy of 32 bytes (two b128 ops) from global to LDS, tracked by ASYNCcnt.
// INST_OFFSET applies to both LDS and global addresses, so offset:16 covers the
// second 16-byte half with the same operands.
__device__ __forceinline__ void async_copy32(unsigned lds_off, unsigned long long gaddr) {
    asm volatile("global_load_async_to_lds_b128 %0, %1, off\n\t"
                 "global_load_async_to_lds_b128 %0, %1, off offset:16"
                 :: "v"(lds_off), "v"(gaddr) : "memory");
}

// LDS byte offset of a __shared__ pointer (flat LDS addr low 32 bits == offset).
__device__ __forceinline__ unsigned lds_offset(const void* p) {
    return (unsigned)(unsigned long long)p;
}

// ---------------------------------------------------------------------------
// Weight transpose + fp32 -> bf16.  W: [K][N] fp32  ->  WT: [N][K] bf16
// ---------------------------------------------------------------------------
__global__ void transpose_f32_to_bf16(const float* __restrict__ W,
                                      unsigned short* __restrict__ WT,
                                      int K, int N)
{
    long i = (long)blockIdx.x * blockDim.x + threadIdx.x;
    if (i >= (long)K * N) return;
    int n = (int)(i % N);
    int k = (int)(i / N);
    WT[(size_t)n * K + k] = f2bf(W[i]);
}

// ---------------------------------------------------------------------------
// Embedding gather: emb_bf16[b*S+s][h] = bf16(E[inputs[b*S+s]][h])
// ---------------------------------------------------------------------------
__global__ void embed_gather(const int* __restrict__ inputs,
                             const float* __restrict__ E,
                             unsigned short* __restrict__ emb)
{
    long i = (long)blockIdx.x * blockDim.x + threadIdx.x;
    if (i >= (long)BS_ * H_) return;
    int  h   = (int)(i % H_);
    long row = i / H_;
    int  tok = inputs[row];
    emb[i] = f2bf(E[(size_t)tok * H_ + h]);
}

__global__ void init_h(const float* __restrict__ h0,
                       float* __restrict__ hf,
                       unsigned short* __restrict__ hb)
{
    int i = blockIdx.x * blockDim.x + threadIdx.x;
    if (i >= B_ * H_) return;
    float v = h0[i];
    hf[i] = v;
    hb[i] = f2bf(v);
}

// ---------------------------------------------------------------------------
// Tiled bf16 WMMA GEMM:  C[M][N] = A[M][K] * BT[N][K]^T + bias
// Block tile 128x128, 256 threads = 8 waves (4 M x 2 N); each wave owns a
// 32x64 patch = 2x4 16x16 WMMA tiles.  K in chunks of 32, double-buffered in
// LDS via GLOBAL_LOAD_ASYNC_TO_LDS_B128 (ASYNCcnt).  Each thread moves one
// 32B segment of A and one of B per chunk (4 async ops).
// Requires M%128==0, N%128==0, K%32==0.
// ---------------------------------------------------------------------------
__global__ __launch_bounds__(256) void gemm_bf16(
    const unsigned short* __restrict__ A,    // [M][K] bf16
    const unsigned short* __restrict__ BT,   // [N][K] bf16
    const float* __restrict__ bias,          // [N] or nullptr
    float* __restrict__ C,                   // [M][N] fp32
    int M, int N, int K)
{
    __shared__ __align__(16) unsigned short As[2][128 * 32];
    __shared__ __align__(16) unsigned short Bs[2][128 * 32];

    const int n0   = blockIdx.x * 128;
    const int m0   = blockIdx.y * 128;
    const int tid  = threadIdx.x;
    const int lane = tid & 31;
    const int w    = tid >> 5;
    const int wm   = w >> 1;       // 0..3 -> 32 rows each
    const int wn   = w & 1;        // 0..1 -> 64 cols each
    const int lr   = lane & 15;
    const int hi   = lane >> 4;

    v8f acc[2][4];
    for (int i = 0; i < 2; i++)
        for (int j = 0; j < 4; j++)
            acc[i][j] = (v8f){0.f,0.f,0.f,0.f,0.f,0.f,0.f,0.f};

    // Staging assignment: thread t moves 32B of A row (t>>1) and 32B of BT row (t>>1).
    const int srow  = tid >> 1;          // 0..127
    const int shalf = (tid & 1) * 16;    // element offset 0 or 16

    const unsigned long long gA = (unsigned long long)(A  + (size_t)(m0 + srow) * K + shalf);
    const unsigned long long gB = (unsigned long long)(BT + (size_t)(n0 + srow) * K + shalf);
    const unsigned ldsA[2] = { lds_offset(&As[0][srow * 32 + shalf]),
                               lds_offset(&As[1][srow * 32 + shalf]) };
    const unsigned ldsB[2] = { lds_offset(&Bs[0][srow * 32 + shalf]),
                               lds_offset(&Bs[1][srow * 32 + shalf]) };

    // Prologue: stage chunk 0 into buffer 0.
    async_copy32(ldsA[0], gA);
    async_copy32(ldsB[0], gB);

    for (int k0 = 0; k0 < K; k0 += 32) {
        const int buf = (k0 >> 5) & 1;
        if (k0 + 32 < K) {
            // Issue chunk k0+32 into the other buffer, then wait for chunk k0
            // (async loads complete in order; <=4 leaves only the new 4 in flight).
            const unsigned long long off = (unsigned long long)(k0 + 32) * 2ull;
            async_copy32(ldsA[buf ^ 1], gA + off);
            async_copy32(ldsB[buf ^ 1], gB + off);
            asm volatile("s_wait_asynccnt 0x4" ::: "memory");
        } else {
            asm volatile("s_wait_asynccnt 0x0" ::: "memory");
        }
        __syncthreads();

        // Fragment loads per ISA 7.12.2 layouts.
        Frag a[2], b[4];
        for (int mt = 0; mt < 2; mt++) {
            const unsigned short* p = &As[buf][(wm * 32 + mt * 16 + lr) * 32 + hi * 8];
            a[mt].q[0] = *(const uint4*)p;          // K = hi*8 .. +7
            a[mt].q[1] = *(const uint4*)(p + 16);   // K = hi*8+16 .. +23
        }
        for (int nt = 0; nt < 4; nt++) {
            const unsigned short* p = &Bs[buf][(wn * 64 + nt * 16 + lr) * 32 + hi * 8];
            b[nt].q[0] = *(const uint4*)p;          // K = hi*8 .. +7
            b[nt].q[1] = *(const uint4*)(p + 16);   // K = hi*8+16 .. +23
        }

        for (int mt = 0; mt < 2; mt++)
            for (int nt = 0; nt < 4; nt++)
                acc[mt][nt] = __builtin_amdgcn_wmma_f32_16x16x32_bf16(
                    false, a[mt].v, false, b[nt].v, (short)0, acc[mt][nt], false, false);
        __syncthreads();   // protect buffer 'buf' before it is re-filled
    }

    // Epilogue: C layout lane l, VGPR j -> (M = (l>>4)*8 + j, N = l&15)
    for (int mt = 0; mt < 2; mt++) {
        for (int nt = 0; nt < 4; nt++) {
            const int n  = n0 + wn * 64 + nt * 16 + lr;
            const float bv = bias ? bias[n] : 0.f;
            for (int j = 0; j < 8; j++) {
                const int m = m0 + wm * 32 + mt * 16 + hi * 8 + j;
                C[(size_t)m * N + n] = acc[mt][nt][j] + bv;
            }
        }
    }
}

// ---------------------------------------------------------------------------
// One GRU time step.  Grid = H/16 = 64 blocks of 128 threads (4 waves).
// The K=1024 contraction is split 4 ways across the waves (256 each); partial
// 32x16 WMMA accumulators for the 3 gates are reduced through LDS, then the
// fused sigmoid/tanh gate math runs in fp32.
// ---------------------------------------------------------------------------
__global__ __launch_bounds__(128) void gru_step(
    const unsigned short* __restrict__ hb,    // h_{t-1} bf16 [B][H]
    const float* __restrict__ hf,             // h_{t-1} fp32 [B][H]
    const unsigned short* __restrict__ WhrT,  // [H][H] bf16 (N-major)
    const unsigned short* __restrict__ WhzT,
    const unsigned short* __restrict__ WhgT,
    const float* __restrict__ bhr,
    const float* __restrict__ bhz,
    const float* __restrict__ bhg,
    const float* __restrict__ xr,             // [B*S][H] fp32
    const float* __restrict__ xz,
    const float* __restrict__ xg,
    int t,
    float* __restrict__ hf_out,
    unsigned short* __restrict__ hb_out,
    unsigned short* __restrict__ hidb)        // [B*S][H] bf16 hiddens
{
    __shared__ float red[4][3][32][16];       // [wave][gate][batch][col] = 24KB

    const int n0   = blockIdx.x * 16;
    const int tid  = threadIdx.x;
    const int lane = tid & 31;
    const int w    = tid >> 5;                // 0..3: K-slice
    const int lr   = lane & 15;
    const int hi   = lane >> 4;
    const int kbeg = w * (H_ / 4);
    const int kend = kbeg + (H_ / 4);

    v8f cR[2], cZ[2], cG[2];
    for (int mt = 0; mt < 2; mt++) {
        cR[mt] = (v8f){0.f,0.f,0.f,0.f,0.f,0.f,0.f,0.f};
        cZ[mt] = cR[mt];
        cG[mt] = cR[mt];
    }

    for (int k0 = kbeg; k0 < kend; k0 += 32) {
        Frag a[2];
        for (int mt = 0; mt < 2; mt++) {
            const unsigned short* p = hb + (size_t)(mt * 16 + lr) * H_ + k0 + hi * 8;
            a[mt].q[0] = *(const uint4*)p;
            a[mt].q[1] = *(const uint4*)(p + 16);
        }
        const size_t bo = (size_t)(n0 + lr) * H_ + k0 + hi * 8;
        Frag br, bz, bg;
        br.q[0] = *(const uint4*)(WhrT + bo); br.q[1] = *(const uint4*)(WhrT + bo + 16);
        bz.q[0] = *(const uint4*)(WhzT + bo); bz.q[1] = *(const uint4*)(WhzT + bo + 16);
        bg.q[0] = *(const uint4*)(WhgT + bo); bg.q[1] = *(const uint4*)(WhgT + bo + 16);

        for (int mt = 0; mt < 2; mt++) {
            cR[mt] = __builtin_amdgcn_wmma_f32_16x16x32_bf16(false, a[mt].v, false, br.v, (short)0, cR[mt], false, false);
            cZ[mt] = __builtin_amdgcn_wmma_f32_16x16x32_bf16(false, a[mt].v, false, bz.v, (short)0, cZ[mt], false, false);
            cG[mt] = __builtin_amdgcn_wmma_f32_16x16x32_bf16(false, a[mt].v, false, bg.v, (short)0, cG[mt], false, false);
        }
    }

    // Spill partial accumulators: C frag lane l, VGPR j -> row (l>>4)*8+j, col l&15.
    for (int mt = 0; mt < 2; mt++) {
        for (int j = 0; j < 8; j++) {
            const int row = mt * 16 + hi * 8 + j;
            red[w][0][row][lr] = cR[mt][j];
            red[w][1][row][lr] = cZ[mt][j];
            red[w][2][row][lr] = cG[mt][j];
        }
    }
    __syncthreads();

    // Reduce over the 4 K-slices and apply the gate nonlinearity.
    for (int e = tid; e < B_ * 16; e += 128) {
        const int b   = e >> 4;
        const int col = e & 15;
        const int n   = n0 + col;
        float sR = 0.f, sZ = 0.f, sG = 0.f;
        for (int ww = 0; ww < 4; ww++) {
            sR += red[ww][0][b][col];
            sZ += red[ww][1][b][col];
            sG += red[ww][2][b][col];
        }
        const size_t xrow = ((size_t)b * S_ + t) * H_ + n;
        const float r  = sigmoidf_(xr[xrow] + sR + bhr[n]);
        const float z  = sigmoidf_(xz[xrow] + sZ + bhz[n]);
        const float g  = tanhf(xg[xrow] + r * (sG + bhg[n]));
        const float hn = (1.f - z) * g + z * hf[(size_t)b * H_ + n];
        hf_out[(size_t)b * H_ + n] = hn;
        const unsigned short h16 = f2bf(hn);
        hb_out[(size_t)b * H_ + n] = h16;
        hidb[xrow] = h16;
    }
}

// ---------------------------------------------------------------------------
extern "C" void kernel_launch(void* const* d_in, const int* in_sizes, int n_in,
                              void* d_out, int out_size, void* d_ws, size_t ws_size,
                              hipStream_t stream)
{
    (void)in_sizes; (void)n_in; (void)out_size; (void)ws_size;

    const int*   inputs      = (const int*)  d_in[0];
    // d_in[1] = annotations (unused by reference)
    const float* hidden_init = (const float*)d_in[2];
    const float* E    = (const float*)d_in[3];
    const float* Wxr  = (const float*)d_in[4];
    const float* bxr  = (const float*)d_in[5];
    const float* Wxz  = (const float*)d_in[6];
    const float* bxz  = (const float*)d_in[7];
    const float* Wxg  = (const float*)d_in[8];
    const float* bxg  = (const float*)d_in[9];
    const float* Whr  = (const float*)d_in[10];
    const float* bhr  = (const float*)d_in[11];
    const float* Whz  = (const float*)d_in[12];
    const float* bhz  = (const float*)d_in[13];
    const float* Whg  = (const float*)d_in[14];
    const float* bhg  = (const float*)d_in[15];
    const float* Wout = (const float*)d_in[16];
    const float* bout = (const float*)d_in[17];
    float* out = (float*)d_out;

    char* ws = (char*)d_ws;
    size_t o = 0;
    auto take = [&](size_t bytes) -> char* {
        size_t r = (o + 255) & ~(size_t)255;
        o = r + bytes;
        return ws + r;
    };

    unsigned short* wxrT  = (unsigned short*)take((size_t)H_ * H_ * 2);
    unsigned short* wxzT  = (unsigned short*)take((size_t)H_ * H_ * 2);
    unsigned short* wxgT  = (unsigned short*)take((size_t)H_ * H_ * 2);
    unsigned short* whrT  = (unsigned short*)take((size_t)H_ * H_ * 2);
    unsigned short* whzT  = (unsigned short*)take((size_t)H_ * H_ * 2);
    unsigned short* whgT  = (unsigned short*)take((size_t)H_ * H_ * 2);
    unsigned short* woutT = (unsigned short*)take((size_t)V_ * H_ * 2);
    unsigned short* emb   = (unsigned short*)take((size_t)BS_ * H_ * 2);
    float*          xr    = (float*)take((size_t)BS_ * H_ * 4);
    float*          xz    = (float*)take((size_t)BS_ * H_ * 4);
    float*          xg    = (float*)take((size_t)BS_ * H_ * 4);
    unsigned short* hidb  = (unsigned short*)take((size_t)BS_ * H_ * 2);
    float*          hf0   = (float*)take((size_t)B_ * H_ * 4);
    float*          hf1   = (float*)take((size_t)B_ * H_ * 4);
    unsigned short* hb0   = (unsigned short*)take((size_t)B_ * H_ * 2);
    unsigned short* hb1   = (unsigned short*)take((size_t)B_ * H_ * 2);

    // 1) Weight conversion/transposition to bf16 [N][K].
    {
        long nHH = (long)H_ * H_;
        int  gHH = (int)((nHH + 255) / 256);
        transpose_f32_to_bf16<<<gHH, 256, 0, stream>>>(Wxr, wxrT, H_, H_);
        transpose_f32_to_bf16<<<gHH, 256, 0, stream>>>(Wxz, wxzT, H_, H_);
        transpose_f32_to_bf16<<<gHH, 256, 0, stream>>>(Wxg, wxgT, H_, H_);
        transpose_f32_to_bf16<<<gHH, 256, 0, stream>>>(Whr, whrT, H_, H_);
        transpose_f32_to_bf16<<<gHH, 256, 0, stream>>>(Whz, whzT, H_, H_);
        transpose_f32_to_bf16<<<gHH, 256, 0, stream>>>(Whg, whgT, H_, H_);
        long nHV = (long)H_ * V_;
        transpose_f32_to_bf16<<<(int)((nHV + 255) / 256), 256, 0, stream>>>(Wout, woutT, H_, V_);
    }

    // 2) Embedding gather to bf16.
    {
        long n = (long)BS_ * H_;
        embed_gather<<<(int)((n + 255) / 256), 256, 0, stream>>>(inputs, E, emb);
    }

    // 3) Input projections xr/xz/xg = emb @ Wx* + bx*  (M=4096, N=1024, K=1024).
    {
        dim3 grid(H_ / 128, BS_ / 128);
        gemm_bf16<<<grid, 256, 0, stream>>>(emb, wxrT, bxr, xr, BS_, H_, H_);
        gemm_bf16<<<grid, 256, 0, stream>>>(emb, wxzT, bxz, xz, BS_, H_, H_);
        gemm_bf16<<<grid, 256, 0, stream>>>(emb, wxgT, bxg, xg, BS_, H_, H_);
    }

    // 4) Hidden-state init (fp32 + bf16 copies).
    init_h<<<(B_ * H_ + 255) / 256, 256, 0, stream>>>(hidden_init, hf0, hb0);

    // 5) Sequential recurrence: 128 dependent step launches, ping-pong h buffers.
    for (int t = 0; t < S_; t++) {
        const int in  = t & 1;
        float*          hfi = in ? hf1 : hf0;
        float*          hfo = in ? hf0 : hf1;
        unsigned short* hbi = in ? hb1 : hb0;
        unsigned short* hbo = in ? hb0 : hb1;
        gru_step<<<H_ / 16, 128, 0, stream>>>(hbi, hfi, whrT, whzT, whgT,
                                              bhr, bhz, bhg, xr, xz, xg, t,
                                              hfo, hbo, hidb);
    }

    // 6) Logits: out = hiddens @ Wout + bout  (M=4096, N=32000, K=1024).
    {
        dim3 grid(V_ / 128, BS_ / 128);
        gemm_bf16<<<grid, 256, 0, stream>>>(hidb, woutT, bout, out, BS_, V_, H_);
    }
}
